// BilinearCategoricalNet_60816736912015
// MI455X (gfx1250) — matrix operand order repaired and checked by reading, not compiled, not run on hardware
//
#include <hip/hip_runtime.h>

// ---------------------------------------------------------------------------
// BilinearCategoricalNet on MI455X (gfx1250, wave32, WMMA).
//
// Roofline: ~326 GFLOP vs ~120 MB unique data => compute bound by ~3 orders
// of magnitude. Use v_wmma_f32_16x16x32_bf16 (K=32) instead of the fp32
// 16x16x4 op (8x matrix throughput), fp32 accumulation. All bf16 operands
// (~107 MB) fit in the 192 MB L2, so fragment loads go straight to global
// memory (L2-resident) without LDS staging.
//
// v2: MLL kernel (84% of FLOPs) widened to 32 rows/wave -> 16 WMMA per
// 24 b128 loads (was 8 per 20), halving VMEM issue pressure per FLOP.
// ---------------------------------------------------------------------------

typedef __attribute__((ext_vector_type(16))) __bf16 v16bf;
typedef __attribute__((ext_vector_type(8)))  float  v8f;

union FragU { v16bf v; uint4 q[2]; };

__device__ __forceinline__ unsigned short f2bf(float f) {
    unsigned int u = __float_as_uint(f);
    return (unsigned short)((u + 0x7FFFu + ((u >> 16) & 1u)) >> 16);
}

// A-matrix fragment (16x32, MxK), 16-bit elements, wave32 layout:
// lane = 16*c + m ; c=0 holds K = k0+[0..7] and k0+16+[0..7],
//                   c=1 holds K = k0+8+[0..7] and k0+24+[0..7].
__device__ __forceinline__ v16bf load_a_frag(const unsigned short* __restrict__ base,
                                             int ld, int row, int k0, int c) {
    const unsigned short* p = base + (size_t)row * ld + k0 + 8 * c;
    FragU u;
    u.q[0] = *reinterpret_cast<const uint4*>(p);        // K = k0+8c   .. +7
    u.q[1] = *reinterpret_cast<const uint4*>(p + 16);   // K = k0+8c+16.. +7
    return u.v;
}

// B-matrix fragment (32x16, KxN). Column n of B == row n of W[N,K] row-major.
// lane = 16*c + n holds K = k0 + 8*c + [0..7] and k0 + 8*c + 16 + [0..7].
__device__ __forceinline__ v16bf load_b_frag(const unsigned short* __restrict__ base,
                                             int ld, int row, int k0, int c) {
    const unsigned short* p = base + (size_t)row * ld + k0 + 8 * c;
    FragU u;
    u.q[0] = *reinterpret_cast<const uint4*>(p);
    u.q[1] = *reinterpret_cast<const uint4*>(p + 16);
    return u.v;
}

__device__ __forceinline__ v8f wmma_bf16(v16bf a, v16bf b, v8f c) {
    return __builtin_amdgcn_wmma_f32_16x16x32_bf16(
        /*neg_a=*/false, a, /*neg_b=*/false, b,
        /*c_mod=*/(short)0, c, /*reuse_a=*/false, /*reuse_b=*/false);
}

// ---------------------------------------------------------------------------
// fp32 -> bf16 (RNE), 4 elements / thread. n is a multiple of 4 for all arrays.
// ---------------------------------------------------------------------------
__global__ void cvt_f32_bf16(const float* __restrict__ in,
                             unsigned short* __restrict__ out, int n) {
    int i = (blockIdx.x * blockDim.x + threadIdx.x) * 4;
    if (i + 3 < n) {
        float4 f = *reinterpret_cast<const float4*>(in + i);
        ushort4 o;
        o.x = f2bf(f.x); o.y = f2bf(f.y); o.z = f2bf(f.z); o.w = f2bf(f.w);
        *reinterpret_cast<ushort4*>(out + i) = o;
    }
}

// ---------------------------------------------------------------------------
// O[M,N](bf16) = relu(A[M,K](bf16) @ W[N,K](bf16)^T + bias[N](f32))
// Block: 256 threads = 8 waves; block tile 128x128; wave tile 64x32
// (4 M-subtiles x 2 N-subtiles => 8 v8f accumulators / wave).
// Grid: (M/128, N/128). K % 32 == 0.
// ---------------------------------------------------------------------------
__global__ __launch_bounds__(256)
void gemm_bias_relu(const unsigned short* __restrict__ A,
                    const unsigned short* __restrict__ W,
                    const float* __restrict__ bias,
                    unsigned short* __restrict__ O,
                    int M, int N, int K) {
    const int lane = threadIdx.x & 31;
    const int wave = threadIdx.x >> 5;   // 0..7
    const int c    = lane >> 4;          // half-wave: 0/1
    const int r    = lane & 15;
    const int wm   = wave >> 2;          // 0..1 -> 64-row slab
    const int wn   = wave & 3;           // 0..3 -> 32-col slab
    const int row0 = blockIdx.x * 128 + wm * 64;
    const int col0 = blockIdx.y * 128 + wn * 32;

    v8f acc[4][2];
#pragma unroll
    for (int mi = 0; mi < 4; ++mi)
#pragma unroll
        for (int ni = 0; ni < 2; ++ni) { v8f z = {}; acc[mi][ni] = z; }

#pragma unroll 2
    for (int k0 = 0; k0 < K; k0 += 32) {
        v16bf a[4], b[2];
#pragma unroll
        for (int mi = 0; mi < 4; ++mi)
            a[mi] = load_a_frag(A, K, row0 + 16 * mi + r, k0, c);
#pragma unroll
        for (int ni = 0; ni < 2; ++ni)
            b[ni] = load_b_frag(W, K, col0 + 16 * ni + r, k0, c);
#pragma unroll
        for (int mi = 0; mi < 4; ++mi)
#pragma unroll
            for (int ni = 0; ni < 2; ++ni)
                acc[mi][ni] = wmma_bf16(a[mi], b[ni], acc[mi][ni]);
    }

    // C/D layout: VGPR j of lane (16c+n) holds element (row = j+8c, col = n).
#pragma unroll
    for (int ni = 0; ni < 2; ++ni) {
        const int col = col0 + 16 * ni + r;
        const float bc = bias[col];
#pragma unroll
        for (int mi = 0; mi < 4; ++mi) {
            const int rbase = row0 + 16 * mi + 8 * c;
#pragma unroll
            for (int j = 0; j < 8; ++j) {
                float v = acc[mi][ni][j] + bc;
                v = v > 0.0f ? v : 0.0f;
                O[(size_t)(rbase + j) * N + col] = f2bf(v);
            }
        }
    }
}

// ---------------------------------------------------------------------------
// Fused MLL + bilinear:
//   logits[b,o] = sum_p (h1[b,:]·W1[o,p,:] + B1[o,p]) * (h2[b,:]·W2[o,p,:] + B2[o,p])
// Block = (256-row tile, one o); 8 waves; each wave: 32 rows x 64 p-cols for
// BOTH products (2 x 4 accumulators x 2 matrices = 16 v8f accs = 128 VGPRs).
// Per K-step: 16 WMMA vs 24 b128 loads (B-frags amortized over 2 M-subtiles).
// p-reduction via __shfl_xor inside 16-lane halves (matches C-layout column
// striping), one f32 store per row. Grid: (M/256, 128).
// ---------------------------------------------------------------------------
__global__ __launch_bounds__(256)
void mll_bilinear(const unsigned short* __restrict__ H1,
                  const unsigned short* __restrict__ H2,
                  const unsigned short* __restrict__ W1,
                  const unsigned short* __restrict__ W2,
                  const float* __restrict__ B1,
                  const float* __restrict__ B2,
                  float* __restrict__ out,
                  int M, int K, int NO) {
    const int lane = threadIdx.x & 31;
    const int wave = threadIdx.x >> 5;   // 0..7
    const int c    = lane >> 4;
    const int r    = lane & 15;
    const int row0 = blockIdx.x * 256 + wave * 32;
    const int o    = blockIdx.y;

    const unsigned short* w1 = W1 + (size_t)o * 64 * K;
    const unsigned short* w2 = W2 + (size_t)o * 64 * K;

    v8f acc1[2][4], acc2[2][4];
#pragma unroll
    for (int m = 0; m < 2; ++m)
#pragma unroll
        for (int t = 0; t < 4; ++t) { v8f z = {}; acc1[m][t] = z; acc2[m][t] = z; }

#pragma unroll 2
    for (int k0 = 0; k0 < K; k0 += 32) {
        v16bf a1[2], a2[2];
#pragma unroll
        for (int m = 0; m < 2; ++m) {
            a1[m] = load_a_frag(H1, K, row0 + 16 * m + r, k0, c);
            a2[m] = load_a_frag(H2, K, row0 + 16 * m + r, k0, c);
        }
#pragma unroll
        for (int t = 0; t < 4; ++t) {
            v16bf b1 = load_b_frag(w1, K, 16 * t + r, k0, c);
            v16bf b2 = load_b_frag(w2, K, 16 * t + r, k0, c);
#pragma unroll
            for (int m = 0; m < 2; ++m) {
                acc1[m][t] = wmma_bf16(a1[m], b1, acc1[m][t]);
                acc2[m][t] = wmma_bf16(a2[m], b2, acc2[m][t]);
            }
        }
    }

    // lane (16c+n), VGPR j of tile t == element (row = 16m + j + 8c, p = 16t+n)
    float s[2][8];
#pragma unroll
    for (int m = 0; m < 2; ++m)
#pragma unroll
        for (int j = 0; j < 8; ++j) s[m][j] = 0.0f;
#pragma unroll
    for (int t = 0; t < 4; ++t) {
        const float bb1 = B1[o * 64 + 16 * t + r];
        const float bb2 = B2[o * 64 + 16 * t + r];
#pragma unroll
        for (int m = 0; m < 2; ++m)
#pragma unroll
            for (int j = 0; j < 8; ++j)
                s[m][j] += (acc1[m][t][j] + bb1) * (acc2[m][t][j] + bb2);
    }
    // reduce over the 16 p-lanes of each half-wave (rows are per-half)
#pragma unroll
    for (int off = 1; off < 16; off <<= 1)
#pragma unroll
        for (int m = 0; m < 2; ++m)
#pragma unroll
            for (int j = 0; j < 8; ++j)
                s[m][j] += __shfl_xor(s[m][j], off, 32);

    if (r == 0) {
#pragma unroll
        for (int m = 0; m < 2; ++m)
#pragma unroll
            for (int j = 0; j < 8; ++j)
                out[(size_t)(row0 + 16 * m + j + 8 * c) * NO + o] = s[m][j];
    }
}

// ---------------------------------------------------------------------------
// Launch
// ---------------------------------------------------------------------------
extern "C" void kernel_launch(void* const* d_in, const int* in_sizes, int n_in,
                              void* d_out, int out_size, void* d_ws, size_t ws_size,
                              hipStream_t stream) {
    constexpr int B  = 8192, NI = 512, H = 1024, NO = 128;

    const float* x1    = (const float*)d_in[0];
    const float* x2    = (const float*)d_in[1];
    const float* m1w1  = (const float*)d_in[2];
    const float* m1b1  = (const float*)d_in[3];
    const float* m1w2  = (const float*)d_in[4];
    const float* m1b2  = (const float*)d_in[5];
    const float* m2w1  = (const float*)d_in[6];
    const float* m2b1  = (const float*)d_in[7];
    const float* m2w2  = (const float*)d_in[8];
    const float* m2b2  = (const float*)d_in[9];
    const float* g1w   = (const float*)d_in[10];
    const float* g1b   = (const float*)d_in[11];
    const float* g2w   = (const float*)d_in[12];
    const float* g2b   = (const float*)d_in[13];

    unsigned short* ws = (unsigned short*)d_ws;
    size_t off = 0;
    auto carve = [&](size_t n) { unsigned short* p = ws + off; off += n; return p; };

    unsigned short* xb1  = carve((size_t)B * NI);     // 4.2M
    unsigned short* xb2  = carve((size_t)B * NI);
    unsigned short* w11  = carve((size_t)H * NI);
    unsigned short* w12  = carve((size_t)H * H);
    unsigned short* w21  = carve((size_t)H * NI);
    unsigned short* w22  = carve((size_t)H * H);
    unsigned short* g1wb = carve((size_t)NO * 64 * H);
    unsigned short* g2wb = carve((size_t)NO * 64 * H);
    unsigned short* bufA = carve((size_t)B * H);      // hidden activations (reused)
    unsigned short* h1   = carve((size_t)B * H);
    unsigned short* h2   = carve((size_t)B * H);
    // total: ~53.5M ushorts (~107 MB) <= ws_size

    auto cvt = [&](const float* in, unsigned short* o, size_t n) {
        int blocks = (int)((n / 4 + 255) / 256);
        cvt_f32_bf16<<<blocks, 256, 0, stream>>>(in, o, (int)n);
    };
    cvt(x1,   xb1,  (size_t)B * NI);
    cvt(x2,   xb2,  (size_t)B * NI);
    cvt(m1w1, w11,  (size_t)H * NI);
    cvt(m1w2, w12,  (size_t)H * H);
    cvt(m2w1, w21,  (size_t)H * NI);
    cvt(m2w2, w22,  (size_t)H * H);
    cvt(g1w,  g1wb, (size_t)NO * 64 * H);
    cvt(g2w,  g2wb, (size_t)NO * 64 * H);

    dim3 blk(256);
    dim3 gmlp1(B / 128, H / 128);   // 64 x 8
    // MLP 1
    gemm_bias_relu<<<gmlp1, blk, 0, stream>>>(xb1,  w11, m1b1, bufA, B, H, NI);
    gemm_bias_relu<<<gmlp1, blk, 0, stream>>>(bufA, w12, m1b2, h1,   B, H, H);
    // MLP 2
    gemm_bias_relu<<<gmlp1, blk, 0, stream>>>(xb2,  w21, m2b1, bufA, B, H, NI);
    gemm_bias_relu<<<gmlp1, blk, 0, stream>>>(bufA, w22, m2b2, h2,   B, H, H);
    // Fused multilinear + bilinear contraction
    dim3 gmll(B / 256, NO);         // 32 x 128
    mll_bilinear<<<gmll, blk, 0, stream>>>(h1, h2, g1wb, g2wb, g1b, g2b,
                                           (float*)d_out, B, H, NO);
}
